// AdaptiveEdgeGATLayer_45947560133206
// MI455X (gfx1250) — matrix operand order repaired and checked by reading, not compiled, not run on hardware
//
#include <hip/hip_runtime.h>

#define N_NODES 50000
#define N_EDGES 800000
#define HEADS 4
#define NEG_SLOPE 0.2f

typedef __attribute__((ext_vector_type(16))) __bf16 bf16x16;
typedef __attribute__((ext_vector_type(8)))  float  f32x8;

#define WMMA_BF16(A, B, C) \
  __builtin_amdgcn_wmma_f32_16x16x32_bf16(false, (A), false, (B), (short)0, (C), false, false)

// ---------- fp32 -> bf16 via native hardware convert ----------
__device__ __forceinline__ __bf16 f2bf(float f) { return (__bf16)f; }
__device__ __forceinline__ unsigned short f2bfu(float f) {
  return __builtin_bit_cast(unsigned short, (__bf16)f);
}
__device__ __forceinline__ __bf16 u2bf(unsigned short s) {
  return __builtin_bit_cast(__bf16, s);
}

// ---------- ordered-uint encoding of float for atomicMax (total order) ----------
__device__ __forceinline__ unsigned fenc(float f) {
  unsigned u = __float_as_uint(f);
  return (u & 0x80000000u) ? ~u : (u | 0x80000000u);
}
__device__ __forceinline__ float fdec(unsigned k) {
  return __uint_as_float((k & 0x80000000u) ? (k ^ 0x80000000u) : ~k);
}
#define ENC_NEG_INF 0x007FFFFFu  // fenc(-inf)

// ---------- WMMA fragment loaders (CDNA5 16x16x32 bf16 layouts) ----------
// A fragment from an LDS-staged bf16 tile; rowbase = &tile[m][k0]
__device__ __forceinline__ bf16x16 load_a_lds(const unsigned short* rowbase, int lane) {
  const unsigned short* p = rowbase + (((lane >> 4) & 1) << 3); // +8 for upper half-wave
  bf16x16 a;
#pragma unroll
  for (int j = 0; j < 8; ++j) a[j] = u2bf(p[j]);          // K = kb + j
#pragma unroll
  for (int j = 0; j < 8; ++j) a[8 + j] = u2bf(p[16 + j]); // K = kb + 16 + j
  return a;
}
// B fragment from pre-transposed bf16 weights Wt[N][K] (contiguous K per lane)
__device__ __forceinline__ bf16x16 load_bt(const unsigned short* Wt, int ldk,
                                           int n0, int k0, int lane) {
  int n = lane & 15;
  int kb = (lane >> 4) << 3;
  const unsigned short* p = Wt + (size_t)(n0 + n) * ldk + k0 + kb;
  bf16x16 b;
#pragma unroll
  for (int j = 0; j < 8; ++j) { b[j] = u2bf(p[j]); b[8 + j] = u2bf(p[16 + j]); } // K = kb + j, kb + 16 + j
  return b;
}

// =======================================================================
// P0: transpose + convert one fp32 weight matrix [K,N] -> bf16 [N,K]
// =======================================================================
__global__ void k_prep_w(const float* __restrict__ W, unsigned short* __restrict__ Wt,
                         int K, int N) {
  int i = blockIdx.x * 256 + threadIdx.x;
  if (i >= K * N) return;
  int k = i / N, n = i - k * N;
  Wt[(size_t)n * K + k] = f2bfu(W[i]);
}

// =======================================================================
// K0: init workspace accumulators
// =======================================================================
__global__ void k_init(float* __restrict__ hnode, float* __restrict__ denom,
                       unsigned* __restrict__ menc, unsigned* __restrict__ emax) {
  int i = blockIdx.x * 256 + threadIdx.x; // grid covers N*128 = 6.4M
  hnode[i] = 0.0f;
  emax[i] = ENC_NEG_INF;
  if (i < N_NODES * HEADS) {
    denom[i] = 0.0f;
    menc[i] = ENC_NEG_INF;
  }
}

// =======================================================================
// K1: f_src = X@W_src+b, f_dst = X@W_dst+b  (16 waves: 2 mats x 8 col tiles)
// =======================================================================
__global__ void k_node_proj(const float* __restrict__ X,
                            const unsigned short* __restrict__ WtS, const float* __restrict__ bsrc,
                            const unsigned short* __restrict__ WtD, const float* __restrict__ bdst,
                            float* __restrict__ fsrc, float* __restrict__ fdst) {
  __shared__ unsigned short Xs[16][136]; // 16x128 bf16 node tile, padded
  int tid = threadIdx.x;
  int lane = tid & 31;
  int wave = tid >> 5;
  int row0 = blockIdx.x * 16;

  { // cooperative stage: 512 threads x 4 floats = 16x128
    int m = tid >> 5, k = (tid & 31) * 4;
    const float4 v = *(const float4*)(X + (size_t)(row0 + m) * 128 + k);
    Xs[m][k + 0] = f2bfu(v.x);
    Xs[m][k + 1] = f2bfu(v.y);
    Xs[m][k + 2] = f2bfu(v.z);
    Xs[m][k + 3] = f2bfu(v.w);
  }
  __syncthreads();

  int mat = wave >> 3;
  int n0 = (wave & 7) * 16;
  const unsigned short* Wt = mat ? WtD : WtS;
  const float* bias = mat ? bdst : bsrc;
  float* out = mat ? fdst : fsrc;
  int m = lane & 15;

  f32x8 c = {};
#pragma unroll
  for (int k0 = 0; k0 < 128; k0 += 32) {
    bf16x16 a = load_a_lds(&Xs[m][k0], lane);
    bf16x16 b = load_bt(Wt, 128, n0, k0, lane);
    c = WMMA_BF16(a, b, c);
  }
  int n = lane & 15;
  int mb = (lane >> 4) * 8;
  float bv = bias[n0 + n];
#pragma unroll
  for (int r = 0; r < 8; ++r)
    out[(size_t)(row0 + mb + r) * 128 + n0 + n] = c[r] + bv;
}

// =======================================================================
// K2: edge_emb = elu(edge_feat@W_edge+b) fused with segment-max scatter
// =======================================================================
__global__ void k_edge_enc(const float* __restrict__ EF,
                           const unsigned short* __restrict__ WtE, const float* __restrict__ be,
                           const int* __restrict__ dst, unsigned* __restrict__ emax) {
  __shared__ unsigned short Es[16][72]; // 16x64 bf16 edge tile, padded
  int tid = threadIdx.x;
  int lane = tid & 31;
  int wave = tid >> 5;  // 8 waves = 8 column tiles
  int row0 = blockIdx.x * 16;

  { // cooperative stage: 256 threads x 4 floats = 16x64
    int m = tid >> 4, k = (tid & 15) * 4;
    const float4 v = *(const float4*)(EF + (size_t)(row0 + m) * 64 + k);
    Es[m][k + 0] = f2bfu(v.x);
    Es[m][k + 1] = f2bfu(v.y);
    Es[m][k + 2] = f2bfu(v.z);
    Es[m][k + 3] = f2bfu(v.w);
  }
  __syncthreads();

  int n0 = wave * 16;
  int m = lane & 15;
  f32x8 c = {};
#pragma unroll
  for (int k0 = 0; k0 < 64; k0 += 32) {
    bf16x16 a = load_a_lds(&Es[m][k0], lane);
    bf16x16 b = load_bt(WtE, 64, n0, k0, lane);
    c = WMMA_BF16(a, b, c);
  }
  int n = lane & 15;
  int mb = (lane >> 4) * 8;
  float bv = be[n0 + n];
#pragma unroll
  for (int r = 0; r < 8; ++r) {
    int row = row0 + mb + r;
    float x = c[r] + bv;
    float v = x > 0.0f ? x : (expf(x) - 1.0f); // ELU
    int d = dst[row];
    atomicMax(&emax[(size_t)d * 128 + n0 + n], fenc(v));
  }
}

// =======================================================================
// K3: per-edge GATv2 logits + segment max (one wave per edge)
// =======================================================================
__global__ void k_logits(const float* __restrict__ fsrc, const float* __restrict__ fdst,
                         const int* __restrict__ src, const int* __restrict__ dst,
                         const float* __restrict__ attn,
                         float* __restrict__ lse, unsigned* __restrict__ menc) {
  int lane = threadIdx.x & 31;
  int wave = threadIdx.x >> 5;
  int e = blockIdx.x * 8 + wave;
  int s = src[e], d = dst[e];
  const float4 vs = *(const float4*)(fsrc + (size_t)s * 128 + lane * 4);
  const float4 vd = *(const float4*)(fdst + (size_t)d * 128 + lane * 4);
  const float4 aw = *(const float4*)(attn + lane * 4);
  float x, acc = 0.0f;
  x = vs.x + vd.x; x = x > 0.0f ? x : NEG_SLOPE * x; acc += x * aw.x;
  x = vs.y + vd.y; x = x > 0.0f ? x : NEG_SLOPE * x; acc += x * aw.y;
  x = vs.z + vd.z; x = x > 0.0f ? x : NEG_SLOPE * x; acc += x * aw.z;
  x = vs.w + vd.w; x = x > 0.0f ? x : NEG_SLOPE * x; acc += x * aw.w;
  // reduce over the 8 lanes covering one head (32 dims / 4 per lane)
  acc += __shfl_xor(acc, 1, 32);
  acc += __shfl_xor(acc, 2, 32);
  acc += __shfl_xor(acc, 4, 32);
  if ((lane & 7) == 0) {
    int h = lane >> 3;
    lse[(size_t)e * 4 + h] = acc;
    atomicMax(&menc[d * 4 + h], fenc(acc));
  }
}

// =======================================================================
// K4: ex = exp(logit - m[dst]); denom[dst] += ex   (1 thread per (e,h))
// =======================================================================
__global__ void k_softmax_denom(float* __restrict__ lse, const int* __restrict__ dst,
                                const unsigned* __restrict__ menc,
                                float* __restrict__ denom) {
  int i = blockIdx.x * 256 + threadIdx.x; // covers E*4
  int e = i >> 2, h = i & 3;
  int d = dst[e];
  float m = fdec(menc[d * 4 + h]); // finite: node has this in-edge
  float ex = expf(lse[i] - m);
  lse[i] = ex;
  atomicAdd(&denom[d * 4 + h], ex);
}

// =======================================================================
// K5: a = ex/denom[dst]; h_node[dst] += f_src[src]*a; emit a (wave/edge)
// =======================================================================
__global__ void k_message(const float* __restrict__ fsrc, const float* __restrict__ lse,
                          const float* __restrict__ denom,
                          const int* __restrict__ src, const int* __restrict__ dst,
                          float* __restrict__ hnode, float* __restrict__ a_out) {
  int lane = threadIdx.x & 31;
  int wave = threadIdx.x >> 5;
  int e = blockIdx.x * 8 + wave;
  int s = src[e], d = dst[e];
  int h = lane >> 3;
  float ex = lse[(size_t)e * 4 + h];
  float dn = denom[d * 4 + h];
  float a = ex / fmaxf(dn, 1e-9f);
  if ((lane & 7) == 0) a_out[(size_t)e * 4 + h] = a;
  const float4 v = *(const float4*)(fsrc + (size_t)s * 128 + lane * 4);
  float* ph = hnode + (size_t)d * 128 + lane * 4;
  atomicAdd(ph + 0, v.x * a);
  atomicAdd(ph + 1, v.y * a);
  atomicAdd(ph + 2, v.z * a);
  atomicAdd(ph + 3, v.w * a);
}

// =======================================================================
// K5b: decode edge_max ordered-uint -> float in place (-inf -> 0)
// =======================================================================
__global__ void k_finalize_emax(unsigned* __restrict__ emax) {
  int i = blockIdx.x * 256 + threadIdx.x; // covers N*128
  float f = fdec(emax[i]);
  if (!isfinite(f)) f = 0.0f;
  emax[i] = __float_as_uint(f);
}

// =======================================================================
// K6: gate = sigmoid([h_node|edge_max]@W_gate+b); fuse; h_out = fused@W_out+b
// =======================================================================
__global__ void k_fuse_out(const float* __restrict__ hnode, const float* __restrict__ emaxf,
                           const unsigned short* __restrict__ WtG, const float* __restrict__ bg,
                           const unsigned short* __restrict__ WtO, const float* __restrict__ bo,
                           float* __restrict__ out) {
  __shared__ unsigned short cat[16][264]; // bf16 [h_node | edge_max] tile
  __shared__ unsigned short hf[16][136];  // bf16 h_fused tile
  int tid = threadIdx.x;
  int lane = tid & 31;
  int wave = tid >> 5; // 8 waves = 8 col tiles
  int row0 = blockIdx.x * 16;
  int n0 = wave * 16;
  int m = lane & 15;
  int n = lane & 15;
  int mb = (lane >> 4) * 8;

  { // cooperative stage of concat input: 16x256 = 4096 elems, 16 per thread
    int mr = tid >> 4;                 // 0..15
    int kbase = (tid & 15) * 16;       // 0..240
    const float* hsrc = (kbase < 128)
        ? hnode + (size_t)(row0 + mr) * 128 + kbase
        : emaxf + (size_t)(row0 + mr) * 128 + (kbase - 128);
#pragma unroll
    for (int j = 0; j < 16; j += 4) {
      const float4 v = *(const float4*)(hsrc + j);
      cat[mr][kbase + j + 0] = f2bfu(v.x);
      cat[mr][kbase + j + 1] = f2bfu(v.y);
      cat[mr][kbase + j + 2] = f2bfu(v.z);
      cat[mr][kbase + j + 3] = f2bfu(v.w);
    }
  }
  __syncthreads();

  // ---- GEMM1: gate logits, K = 256 over concat(h_node, edge_max) ----
  f32x8 c = {};
#pragma unroll
  for (int k0 = 0; k0 < 256; k0 += 32) {
    bf16x16 a = load_a_lds(&cat[m][k0], lane);
    bf16x16 b = load_bt(WtG, 256, n0, k0, lane);
    c = WMMA_BF16(a, b, c);
  }
  float bgv = bg[n0 + n];
#pragma unroll
  for (int r = 0; r < 8; ++r) {
    int row = row0 + mb + r;
    float g = 1.0f / (1.0f + expf(-(c[r] + bgv)));
    float hn = hnode[(size_t)row * 128 + n0 + n];
    float em = emaxf[(size_t)row * 128 + n0 + n];
    float fused = g * hn + (1.0f - g) * em;
    hf[mb + r][n0 + n] = f2bfu(fused);
  }
  __syncthreads();

  // ---- GEMM2: h_out = h_fused @ W_out + b_out, K = 128 from LDS ----
  f32x8 c2 = {};
#pragma unroll
  for (int k0 = 0; k0 < 128; k0 += 32) {
    bf16x16 a = load_a_lds(&hf[m][k0], lane);
    bf16x16 b = load_bt(WtO, 128, n0, k0, lane);
    c2 = WMMA_BF16(a, b, c2);
  }
  float bov = bo[n0 + n];
#pragma unroll
  for (int r = 0; r < 8; ++r)
    out[(size_t)(row0 + mb + r) * 128 + n0 + n] = c2[r] + bov;
}

// =======================================================================
extern "C" void kernel_launch(void* const* d_in, const int* in_sizes, int n_in,
                              void* d_out, int out_size, void* d_ws, size_t ws_size,
                              hipStream_t stream) {
  const float* node_feat = (const float*)d_in[0];
  const float* edge_feat = (const float*)d_in[1];
  const int*   src       = (const int*)d_in[2];
  const int*   dst       = (const int*)d_in[3];
  const float* W_src     = (const float*)d_in[4];
  const float* b_src     = (const float*)d_in[5];
  const float* W_dst     = (const float*)d_in[6];
  const float* b_dst     = (const float*)d_in[7];
  const float* attn_w    = (const float*)d_in[8];
  const float* W_edge    = (const float*)d_in[9];
  const float* b_edge    = (const float*)d_in[10];
  const float* W_gate    = (const float*)d_in[11];
  const float* b_gate    = (const float*)d_in[12];
  const float* W_out     = (const float*)d_in[13];
  const float* b_out     = (const float*)d_in[14];

  // workspace layout (floats): ~117 MB
  float* ws = (float*)d_ws;
  float*    f_src = ws;                               // 6,400,000
  float*    f_dst = ws + 6400000;                     // 6,400,000
  float*    lse   = ws + 12800000;                    // 3,200,000 (logits -> ex)
  unsigned* menc  = (unsigned*)(ws + 16000000);       //   200,000
  float*    denom = ws + 16200000;                    //   200,000
  float*    hnode = ws + 16400000;                    // 6,400,000
  unsigned* emax  = (unsigned*)(ws + 22800000);       // 6,400,000 (uint -> float in place)
  unsigned short* wt = (unsigned short*)(ws + 29200000);
  unsigned short* wt_src  = wt;            // 128*128
  unsigned short* wt_dst  = wt + 16384;    // 128*128
  unsigned short* wt_edge = wt + 32768;    // 64*128 -> [128][64]
  unsigned short* wt_gate = wt + 40960;    // 256*128 -> [128][256]
  unsigned short* wt_out  = wt + 73728;    // 128*128

  float* h_out = (float*)d_out;    // N*128
  float* a_out = h_out + 6400000;  // E*4

  // weight prep: transpose + bf16 convert (tiny, stays in L2)
  k_prep_w<<<64, 256, 0, stream>>>(W_src, wt_src, 128, 128);
  k_prep_w<<<64, 256, 0, stream>>>(W_dst, wt_dst, 128, 128);
  k_prep_w<<<32, 256, 0, stream>>>(W_edge, wt_edge, 64, 128);
  k_prep_w<<<128, 256, 0, stream>>>(W_gate, wt_gate, 256, 128);
  k_prep_w<<<64, 256, 0, stream>>>(W_out, wt_out, 128, 128);

  k_init<<<25000, 256, 0, stream>>>(hnode, denom, menc, emax);
  k_node_proj<<<3125, 512, 0, stream>>>(node_feat, wt_src, b_src, wt_dst, b_dst, f_src, f_dst);
  k_edge_enc<<<50000, 256, 0, stream>>>(edge_feat, wt_edge, b_edge, dst, emax);
  k_logits<<<100000, 256, 0, stream>>>(f_src, f_dst, src, dst, attn_w, lse, menc);
  k_softmax_denom<<<12500, 256, 0, stream>>>(lse, dst, menc, denom);
  k_message<<<100000, 256, 0, stream>>>(f_src, lse, denom, src, dst, hnode, a_out);
  k_finalize_emax<<<25000, 256, 0, stream>>>(emax);
  k_fuse_out<<<3125, 256, 0, stream>>>(hnode, (const float*)emax,
                                       wt_gate, b_gate, wt_out, b_out, h_out);
}
